// GIN_76613626626159
// MI455X (gfx1250) — matrix operand order repaired and checked by reading, not compiled, not run on hardware
//
#include <hip/hip_runtime.h>

#define N_NODES    100000
#define N_EDGES    1600000
#define IN_DIM     128
#define HID        64
#define OUT_DIM    5
#define NUM_GRAPHS 128

typedef __bf16 bf16_t;
typedef __attribute__((ext_vector_type(16))) __bf16 bf16x16;
typedef __attribute__((ext_vector_type(8)))  float  v8f;

// ---------------------------------------------------------------------------
// Zero a float buffer (grid-stride)
// ---------------------------------------------------------------------------
__global__ void k_zero(float* __restrict__ p, long long n) {
    long long i = (long long)blockIdx.x * blockDim.x + threadIdx.x;
    long long stride = (long long)gridDim.x * blockDim.x;
    for (; i < n; i += stride) p[i] = 0.0f;
}

// ---------------------------------------------------------------------------
// Pack f32 row-major [K][N] weights -> bf16 column-major [N][K]
// (column-major makes each WMMA B-fragment a single contiguous 32B load)
// ---------------------------------------------------------------------------
__device__ inline void pack_one(const float* __restrict__ W, int K, int N,
                                bf16_t* __restrict__ out, int tid, int nth) {
    for (int idx = tid; idx < K * N; idx += nth) {
        int k = idx / N, n = idx % N;
        out[n * K + k] = (bf16_t)W[idx];
    }
}

__global__ void k_pack_weights(const float* W1a, const float* W1b,
                               const float* W2a, const float* W2b,
                               const float* Wg,
                               bf16_t* p1a, bf16_t* p1b,
                               bf16_t* p2a, bf16_t* p2b, bf16_t* pg) {
    int tid = blockIdx.x * blockDim.x + threadIdx.x;
    int nth = gridDim.x * blockDim.x;
    pack_one(W1a, IN_DIM, HID, p1a, tid, nth);
    pack_one(W1b, HID,   HID, p1b, tid, nth);
    pack_one(W2a, HID,   HID, p2a, tid, nth);
    pack_one(W2b, HID,   HID, p2b, tid, nth);
    pack_one(Wg,  HID,   HID, pg,  tid, nth);
}

// ---------------------------------------------------------------------------
// Edge scatter-add:  agg[dst] += feat[src]   (float4 gather + f32 atomics;
// working set is L2-resident on MI455X so atomics resolve in L2)
// ---------------------------------------------------------------------------
template <int DIM>
__global__ void k_edge_scatter(const float* __restrict__ feat,
                               const long long* __restrict__ src,
                               const long long* __restrict__ dst,
                               float* __restrict__ agg) {
    constexpr int CH = DIM / 4;
    long long t = (long long)blockIdx.x * blockDim.x + threadIdx.x;
    long long e = t / CH;
    int c = (int)(t % CH);
    if (e >= N_EDGES) return;
    long long s = src[e];
    long long d = dst[e];
    float4 v = *reinterpret_cast<const float4*>(feat + s * DIM + c * 4);
    float* o = agg + d * DIM + c * 4;
    atomicAdd(o + 0, v.x);
    atomicAdd(o + 1, v.y);
    atomicAdd(o + 2, v.z);
    atomicAdd(o + 3, v.w);
}

// ---------------------------------------------------------------------------
// A-fragment loader: 16x32 bf16 tile, contiguous k order (same order as the B loads).
// lane (0..31): row m = lane&15, half = lane>>4.
//   elements 0..15 <-> k = kb + 16*half + 0..15
// Optionally fuses the GIN residual add (x + agg) before f32->bf16 round.
// ---------------------------------------------------------------------------
template <bool HAS_A2>
__device__ inline bf16x16 make_a_frag(const float* __restrict__ rowA,
                                      const float* __restrict__ rowA2,
                                      int kb, int half) {
    const int o = kb + 16 * half;
    float4 f0 = *reinterpret_cast<const float4*>(rowA + o);
    float4 f1 = *reinterpret_cast<const float4*>(rowA + o + 4);
    float4 f2 = *reinterpret_cast<const float4*>(rowA + o + 8);
    float4 f3 = *reinterpret_cast<const float4*>(rowA + o + 12);
    if constexpr (HAS_A2) {
        float4 g0 = *reinterpret_cast<const float4*>(rowA2 + o);
        float4 g1 = *reinterpret_cast<const float4*>(rowA2 + o + 4);
        float4 g2 = *reinterpret_cast<const float4*>(rowA2 + o + 8);
        float4 g3 = *reinterpret_cast<const float4*>(rowA2 + o + 12);
        f0.x += g0.x; f0.y += g0.y; f0.z += g0.z; f0.w += g0.w;
        f1.x += g1.x; f1.y += g1.y; f1.z += g1.z; f1.w += g1.w;
        f2.x += g2.x; f2.y += g2.y; f2.z += g2.z; f2.w += g2.w;
        f3.x += g3.x; f3.y += g3.y; f3.z += g3.z; f3.w += g3.w;
    }
    bf16x16 a;
    a[0]  = (bf16_t)f0.x; a[1]  = (bf16_t)f0.y; a[2]  = (bf16_t)f0.z; a[3]  = (bf16_t)f0.w;
    a[4]  = (bf16_t)f1.x; a[5]  = (bf16_t)f1.y; a[6]  = (bf16_t)f1.z; a[7]  = (bf16_t)f1.w;
    a[8]  = (bf16_t)f2.x; a[9]  = (bf16_t)f2.y; a[10] = (bf16_t)f2.z; a[11] = (bf16_t)f2.w;
    a[12] = (bf16_t)f3.x; a[13] = (bf16_t)f3.y; a[14] = (bf16_t)f3.z; a[15] = (bf16_t)f3.w;
    return a;
}

// ---------------------------------------------------------------------------
// WMMA GEMM, N fixed at 64:  out[M][64] = act((A (+A2)) @ W + bias)
// W given as bf16 column-major [64][K]. Block = 8 waves; each wave owns a
// 16-row tile (M = 100000 is a multiple of 16 -> EXEC stays all-ones).
// ---------------------------------------------------------------------------
template <int K, bool HAS_A2, bool RELU>
__global__ void k_gemm_n64(const float* __restrict__ A,
                           const float* __restrict__ A2,
                           const bf16_t* __restrict__ Wcm,
                           const float* __restrict__ bias,
                           float* __restrict__ out, int M) {
    const int wave = threadIdx.x >> 5;
    const int lane = threadIdx.x & 31;
    const int row0 = (blockIdx.x * 8 + wave) * 16;
    if (row0 >= M) return;                       // uniform per wave

    const int m    = lane & 15;
    const int half = lane >> 4;
    const float* rowA  = A + (long long)(row0 + m) * K;
    const float* rowA2 = HAS_A2 ? (A2 + (long long)(row0 + m) * K) : nullptr;

    v8f c0 = {}, c1 = {}, c2 = {}, c3 = {};
#pragma unroll
    for (int kb = 0; kb < K; kb += 32) {
        bf16x16 a = make_a_frag<HAS_A2>(rowA, rowA2, kb, half);
        // B fragment: lane holds column n = nt*16 + (lane&15), k = kb+16*half+i
        const bf16_t* wp = Wcm + (lane & 15) * K + kb + 16 * half;
        bf16x16 b0 = *reinterpret_cast<const bf16x16*>(wp + 0 * 16 * K);
        bf16x16 b1 = *reinterpret_cast<const bf16x16*>(wp + 1 * 16 * K);
        bf16x16 b2 = *reinterpret_cast<const bf16x16*>(wp + 2 * 16 * K);
        bf16x16 b3 = *reinterpret_cast<const bf16x16*>(wp + 3 * 16 * K);
        c0 = __builtin_amdgcn_wmma_f32_16x16x32_bf16(false, a, false, b0, (short)0, c0, false, false);
        c1 = __builtin_amdgcn_wmma_f32_16x16x32_bf16(false, a, false, b1, (short)0, c1, false, false);
        c2 = __builtin_amdgcn_wmma_f32_16x16x32_bf16(false, a, false, b2, (short)0, c2, false, false);
        c3 = __builtin_amdgcn_wmma_f32_16x16x32_bf16(false, a, false, b3, (short)0, c3, false, false);
    }

    const int n0 = lane & 15;
#pragma unroll
    for (int j = 0; j < 8; ++j) {
        int r = row0 + j + 8 * half;             // C/D layout: VGPR j -> row j+8*half
        float* po = out + (long long)r * HID;
        float vals[4] = { c0[j], c1[j], c2[j], c3[j] };
#pragma unroll
        for (int nt = 0; nt < 4; ++nt) {
            int n = nt * 16 + n0;
            float v = vals[nt] + bias[n];
            if (RELU) v = fmaxf(v, 0.0f);
            po[n] = v;
        }
    }
}

// ---------------------------------------------------------------------------
// global_add_pool: g[batch[node]] += h[node]   (8K-float hot set, L2 atomics)
// ---------------------------------------------------------------------------
__global__ void k_pool(const float* __restrict__ h,
                       const long long* __restrict__ batch,
                       float* __restrict__ g) {
    long long t = (long long)blockIdx.x * blockDim.x + threadIdx.x;
    long long node = t >> 4;
    int c = (int)(t & 15);
    if (node >= N_NODES) return;
    long long b = batch[node];
    float4 v = *reinterpret_cast<const float4*>(h + node * HID + c * 4);
    float* o = g + b * HID + c * 4;
    atomicAdd(o + 0, v.x);
    atomicAdd(o + 1, v.y);
    atomicAdd(o + 2, v.z);
    atomicAdd(o + 3, v.w);
}

// ---------------------------------------------------------------------------
// Head: relu(g @ Wg + bg) via WMMA (staged in LDS), then @ Wo + bo (N=5, VALU)
// Single block, 8 waves, 128 rows.
// ---------------------------------------------------------------------------
__global__ void k_head(const float* __restrict__ g,
                       const bf16_t* __restrict__ Wgcm,
                       const float* __restrict__ bg,
                       const float* __restrict__ Wo,
                       const float* __restrict__ bo,
                       float* __restrict__ out) {
    __shared__ float sg[NUM_GRAPHS * HID];

    const int wave = threadIdx.x >> 5;
    const int lane = threadIdx.x & 31;
    const int row0 = wave * 16;
    const int m    = lane & 15;
    const int half = lane >> 4;
    const float* rowA = g + (row0 + m) * HID;

    v8f c0 = {}, c1 = {}, c2 = {}, c3 = {};
#pragma unroll
    for (int kb = 0; kb < HID; kb += 32) {
        bf16x16 a = make_a_frag<false>(rowA, nullptr, kb, half);
        const bf16_t* wp = Wgcm + (lane & 15) * HID + kb + 16 * half;
        bf16x16 b0 = *reinterpret_cast<const bf16x16*>(wp + 0 * 16 * HID);
        bf16x16 b1 = *reinterpret_cast<const bf16x16*>(wp + 1 * 16 * HID);
        bf16x16 b2 = *reinterpret_cast<const bf16x16*>(wp + 2 * 16 * HID);
        bf16x16 b3 = *reinterpret_cast<const bf16x16*>(wp + 3 * 16 * HID);
        c0 = __builtin_amdgcn_wmma_f32_16x16x32_bf16(false, a, false, b0, (short)0, c0, false, false);
        c1 = __builtin_amdgcn_wmma_f32_16x16x32_bf16(false, a, false, b1, (short)0, c1, false, false);
        c2 = __builtin_amdgcn_wmma_f32_16x16x32_bf16(false, a, false, b2, (short)0, c2, false, false);
        c3 = __builtin_amdgcn_wmma_f32_16x16x32_bf16(false, a, false, b3, (short)0, c3, false, false);
    }

    const int n0 = lane & 15;
#pragma unroll
    for (int j = 0; j < 8; ++j) {
        int r = row0 + j + 8 * half;
        float vals[4] = { c0[j], c1[j], c2[j], c3[j] };
#pragma unroll
        for (int nt = 0; nt < 4; ++nt) {
            int n = nt * 16 + n0;
            sg[r * HID + n] = fmaxf(vals[nt] + bg[n], 0.0f);
        }
    }
    __syncthreads();

    for (int idx = threadIdx.x; idx < NUM_GRAPHS * OUT_DIM; idx += blockDim.x) {
        int r = idx / OUT_DIM, c = idx % OUT_DIM;
        float acc = bo[c];
#pragma unroll 8
        for (int k = 0; k < HID; ++k) acc += sg[r * HID + k] * Wo[k * OUT_DIM + c];
        out[idx] = acc;
    }
}

// ---------------------------------------------------------------------------
// Launch
// ---------------------------------------------------------------------------
extern "C" void kernel_launch(void* const* d_in, const int* in_sizes, int n_in,
                              void* d_out, int out_size, void* d_ws, size_t ws_size,
                              hipStream_t stream) {
    (void)in_sizes; (void)n_in; (void)out_size; (void)ws_size;

    const float*     x     = (const float*)d_in[0];
    const long long* ei    = (const long long*)d_in[1];   // int64 (2, E)
    const long long* src   = ei;
    const long long* dst   = ei + N_EDGES;
    const long long* batch = (const long long*)d_in[2];   // int64 (N,)
    const float* W1a = (const float*)d_in[3];  const float* b1a = (const float*)d_in[4];
    const float* W1b = (const float*)d_in[5];  const float* b1b = (const float*)d_in[6];
    const float* W2a = (const float*)d_in[7];  const float* b2a = (const float*)d_in[8];
    const float* W2b = (const float*)d_in[9];  const float* b2b = (const float*)d_in[10];
    const float* Wg  = (const float*)d_in[11]; const float* bg  = (const float*)d_in[12];
    const float* Wo  = (const float*)d_in[13]; const float* bo  = (const float*)d_in[14];
    float* out = (float*)d_out;

    // Workspace layout (floats), ~128.07 MB total:
    float* agg1 = (float*)d_ws;                              // N*128   (agg of conv1)
    float* bufB = agg1 + (size_t)N_NODES * IN_DIM;           // N*64    (t1, later t2)
    float* h1   = bufB + (size_t)N_NODES * HID;              // N*64
    float* bufD = h1   + (size_t)N_NODES * HID;              // N*64    (agg2, later h2)
    float* gbuf = bufD + (size_t)N_NODES * HID;              // 128*64
    bf16_t* p1a = (bf16_t*)(gbuf + NUM_GRAPHS * HID);        // packed bf16 weights
    bf16_t* p1b = p1a + IN_DIM * HID;
    bf16_t* p2a = p1b + HID * HID;
    bf16_t* p2b = p2a + HID * HID;
    bf16_t* pg  = p2b + HID * HID;

    // 0) zero accumulation buffers + pack weights to bf16 column-major
    k_zero<<<2048, 256, 0, stream>>>(agg1, (long long)N_NODES * IN_DIM);
    k_zero<<<1024, 256, 0, stream>>>(bufD, (long long)N_NODES * HID);
    k_zero<<<8,    256, 0, stream>>>(gbuf, (long long)NUM_GRAPHS * HID);
    k_pack_weights<<<32, 256, 0, stream>>>(W1a, W1b, W2a, W2b, Wg, p1a, p1b, p2a, p2b, pg);

    const int gemm_grid = (N_NODES + 127) / 128;             // 8 waves x 16 rows / block

    // 1) conv1: agg1 = segsum(x[src] -> dst); t1 = relu((x+agg1)@W1a+b1a); h1 = relu(t1@W1b+b1b)
    {
        long long th = (long long)N_EDGES * (IN_DIM / 4);
        k_edge_scatter<IN_DIM><<<(unsigned)((th + 255) / 256), 256, 0, stream>>>(x, src, dst, agg1);
    }
    k_gemm_n64<IN_DIM, true,  true><<<gemm_grid, 256, 0, stream>>>(x,    agg1,    p1a, b1a, bufB, N_NODES);
    k_gemm_n64<HID,    false, true><<<gemm_grid, 256, 0, stream>>>(bufB, nullptr, p1b, b1b, h1,   N_NODES);

    // 2) conv2: agg2 = segsum(h1[src] -> dst); t2 = relu((h1+agg2)@W2a+b2a); h2 = relu(t2@W2b+b2b)
    {
        long long th = (long long)N_EDGES * (HID / 4);
        k_edge_scatter<HID><<<(unsigned)((th + 255) / 256), 256, 0, stream>>>(h1, src, dst, bufD);
    }
    k_gemm_n64<HID, true,  true><<<gemm_grid, 256, 0, stream>>>(h1,   bufD,    p2a, b2a, bufB, N_NODES);
    k_gemm_n64<HID, false, true><<<gemm_grid, 256, 0, stream>>>(bufB, nullptr, p2b, b2b, bufD, N_NODES);

    // 3) pool + head
    {
        long long th = (long long)N_NODES * 16;
        k_pool<<<(unsigned)((th + 255) / 256), 256, 0, stream>>>(bufD, batch, gbuf);
    }
    k_head<<<1, 256, 0, stream>>>(gbuf, pg, bg, Wo, bo, out);
}
